// UberCRSN_24902220382715
// MI455X (gfx1250) — matrix operand order repaired and checked by reading, not compiled, not run on hardware
//
#include <hip/hip_runtime.h>

typedef __attribute__((ext_vector_type(16))) __bf16 v16bf;
typedef __attribute__((ext_vector_type(8)))  __bf16 v8bf;
typedef __attribute__((ext_vector_type(8)))  float  v8f;

#define DEV static __device__ __forceinline__

constexpr int   Bsz   = 4096;
constexpr int   Dd    = 128;
constexpr int   Ss    = 24;
constexpr int   DEPTH = 10;
constexpr float EPSV  = 1e-6f;
constexpr float THRv  = 0.99f;
constexpr int   TB    = 64;        // batch rows per block
constexpr int   NBLK  = Bsz / TB;  // 64 blocks
constexpr int   NTH   = 256;       // 8 wave32s
constexpr int   WTILE = Dd * Dd;   // elements per weight tile (16384)

// bf16 weight tiles staged in workspace:
// 0 lwr, 1 lwi, 2 -lwi, 3 vwr, 4 vwi, 5 -vwi, 6 owr, 7 owi, 8 -owi, 9..12 gw_j
constexpr int NWT = 13;

// ---------------- LDS layout (bytes) ----------------
constexpr int OFF_WBUF = 0;                           // 3 x (128x128) bf16 weight slots (Wr, Wi, -Wi)
constexpr int OFF_ACT  = OFF_WBUF + 3 * Dd * Dd * 2;  // 4 x (64x128) bf16 activation slots
constexpr int OFF_FST  = OFF_ACT  + 4 * TB * Dd * 2;  // 3 x (64x128) f32 staging slots
constexpr int OFF_PCUR = OFF_FST  + 3 * TB * Dd * 4;  // 64x24 stack pointer
constexpr int OFF_WM   = OFF_PCUR + TB * Ss * 4;      // 64x24 write-mask
constexpr int OFF_CUM  = OFF_WM   + TB * Ss * 4;      // 64
constexpr int OFF_WROW = OFF_CUM  + TB * 4;           // 64
constexpr int OFF_SC   = OFF_WROW + TB * 4;           // 128
constexpr int OFF_SH   = OFF_SC + Dd * 4;
constexpr int OFF_MB   = OFF_SH + Dd * 4;
constexpr int OFF_GB   = OFF_MB + Dd * 4;
constexpr int OFF_HW   = OFF_GB + Dd * 4;             // 256
constexpr int OFF_SW   = OFF_HW + 2 * Dd * 4;         // 3x256
constexpr int SMEM_SZ  = OFF_SW + 3 * 2 * Dd * 4;     // ~275 KB < 320 KB

DEV unsigned short f2bf(float f) {
  union { float f; unsigned u; } c; c.f = f;
  unsigned u = c.u;
  u += 0x7fffu + ((u >> 16) & 1u);   // round-to-nearest-even
  return (unsigned short)(u >> 16);
}
DEV float bf2f(unsigned short h) {
  union { unsigned u; float f; } c; c.u = ((unsigned)h) << 16;
  return c.f;
}
DEV float sigm(float x) { return 1.0f / (1.0f + __expf(-x)); }

// ---- WMMA fragment helpers (bf16, 16x16x32) ----
DEV v16bf fragA(const unsigned short* base, int lane, int k0) {
  const int r = lane & 15, hi = lane >> 4;
  const v8bf a = *(const v8bf*)(base + r * Dd + k0 + hi * 8);
  const v8bf b = *(const v8bf*)(base + r * Dd + k0 + 16 + hi * 8);
  return __builtin_shufflevector(a, b, 0,1,2,3,4,5,6,7,8,9,10,11,12,13,14,15);
}
DEV v16bf fragB(const unsigned short* w, int lane, int ntile, int k0) {
  const int c = lane & 15, hi = lane >> 4;
  const v8bf a = *(const v8bf*)(w + (ntile * 16 + c) * Dd + k0 + hi * 8);
  const v8bf b = *(const v8bf*)(w + (ntile * 16 + c) * Dd + k0 + 16 + hi * 8);
  return __builtin_shufflevector(a, b, 0,1,2,3,4,5,6,7,8,9,10,11,12,13,14,15);
}
DEV v8f wmma_bf(v16bf a, v16bf b, v8f c) {
  return __builtin_amdgcn_wmma_f32_16x16x32_bf16(false, a, false, b, (short)0, c,
                                                 false, false);
}

// One-time f32 -> bf16 weight conversion into workspace (global).
DEV void conv_ws(unsigned short* dst, const float* __restrict__ src,
                 int stride, int tid) {
  for (int idx = tid; idx < WTILE / 4; idx += NTH) {
    const int r = idx >> 5;
    const int c = (idx & 31) * 4;
    const float4 v = *(const float4*)(src + r * stride + c);
    unsigned long long p =
        (unsigned long long)f2bf(v.x) |
        ((unsigned long long)f2bf(v.y) << 16) |
        ((unsigned long long)f2bf(v.z) << 32) |
        ((unsigned long long)f2bf(v.w) << 48);
    *(unsigned long long*)(dst + r * Dd + c) = p;
  }
}
// Same but writes both +W and -W tiles (bf16 WMMA has no A/B negate modifier).
DEV void conv_ws_pm(unsigned short* dstp, unsigned short* dstn,
                    const float* __restrict__ src, int stride, int tid) {
  for (int idx = tid; idx < WTILE / 4; idx += NTH) {
    const int r = idx >> 5;
    const int c = (idx & 31) * 4;
    const float4 v = *(const float4*)(src + r * stride + c);
    unsigned long long p =
        (unsigned long long)f2bf(v.x) |
        ((unsigned long long)f2bf(v.y) << 16) |
        ((unsigned long long)f2bf(v.z) << 32) |
        ((unsigned long long)f2bf(v.w) << 48);
    *(unsigned long long*)(dstp + r * Dd + c) = p;
    *(unsigned long long*)(dstn + r * Dd + c) = p ^ 0x8000800080008000ull;
  }
}

// Async DMA copy global(bf16 tiles) -> LDS, tracked with ASYNCcnt.
// 16 bytes per lane per issue; caller must async_fence() + __syncthreads().
DEV void async_copy(void* lds_dst, const void* gsrc, int nbytes, int tid) {
  const unsigned lbase = (unsigned)(size_t)lds_dst;   // low 32 bits = LDS offset
  for (int off = tid * 16; off < nbytes; off += NTH * 16) {
    const unsigned l = lbase + off;
    const char* g = (const char*)gsrc + off;
    asm volatile("global_load_async_to_lds_b128 %0, %1, off"
                 :: "v"(l), "v"(g) : "memory");
  }
}
DEV void async_fence() {
  asm volatile("s_wait_asynccnt 0x0" ::: "memory");
}

// Prefetch next stage's bf16 weight tiles (global_prefetch_b8 on gfx1250).
DEV void prefetch_bf(const unsigned short* p, int nelem, int tid) {
  for (int i = tid * 64; i < nelem; i += NTH * 64) __builtin_prefetch(p + i);
}

// Complex GEMM tile: (Or + i*Oi) = (Ar + i*Ai)(64x128) @ (Wr + i*Wi).T(128x128)
// Wni = -Wi staged in LDS. A-fragments hoisted out of the N loop.
DEV void cgemm(const unsigned short* Ar, const unsigned short* Ai,
               const unsigned short* Wr, const unsigned short* Wi,
               const unsigned short* Wni,
               float* Or, float* Oi, int wave, int lane) {
  const int strip = wave >> 1;
  const int nbase = (wave & 1) * 4;
  const unsigned short* abr = Ar + strip * 16 * Dd;
  const unsigned short* abi = Ai + strip * 16 * Dd;
  const int rbase = strip * 16 + 8 * (lane >> 4);
  const int col0  = lane & 15;
  v16bf far_[4], fai_[4];
#pragma unroll
  for (int kk = 0; kk < 4; ++kk) {
    far_[kk] = fragA(abr, lane, kk * 32);
    fai_[kk] = fragA(abi, lane, kk * 32);
  }
  for (int nt = 0; nt < 4; ++nt) {
    const int ntile = nbase + nt;
    v8f accr = {0,0,0,0,0,0,0,0};
    v8f acci = {0,0,0,0,0,0,0,0};
#pragma unroll
    for (int kk = 0; kk < 4; ++kk) {
      const int k0 = kk * 32;
      const v16bf fb_r = fragB(Wr,  lane, ntile, k0);
      const v16bf fb_i = fragB(Wi,  lane, ntile, k0);
      const v16bf fb_n = fragB(Wni, lane, ntile, k0);
      accr = wmma_bf(far_[kk], fb_r, accr);   //  ar*wr
      accr = wmma_bf(fai_[kk], fb_n, accr);   //  ai*(-wi)
      acci = wmma_bf(fai_[kk], fb_r, acci);   //  ai*wr
      acci = wmma_bf(far_[kk], fb_i, acci);   //  ar*wi
    }
    const int col = ntile * 16 + col0;
#pragma unroll
    for (int v = 0; v < 8; ++v) {
      Or[(rbase + v) * Dd + col] = accr[v];
      Oi[(rbase + v) * Dd + col] = acci[v];
    }
  }
}

__global__ __launch_bounds__(NTH, 1)
void ubercrsn_kernel(const float* __restrict__ zr_in, const float* __restrict__ zi_in,
                     const float* __restrict__ mem_in, const float* __restrict__ ptr_in,
                     const float* __restrict__ lwr, const float* __restrict__ lwi,
                     const float* __restrict__ nsc, const float* __restrict__ nsh,
                     const float* __restrict__ mbi,
                     const float* __restrict__ vwr, const float* __restrict__ vwi,
                     const float* __restrict__ owr, const float* __restrict__ owi,
                     const float* __restrict__ hw,  const float* __restrict__ hb,
                     const float* __restrict__ sw,  const float* __restrict__ sb,
                     const float* __restrict__ gw,  const float* __restrict__ gb,
                     float* __restrict__ out,
                     float* __restrict__ wmem,
                     float* __restrict__ waccr, float* __restrict__ wacci,
                     unsigned short* __restrict__ wbf) {
  extern __shared__ unsigned char smem[];
  unsigned short* wbuf = (unsigned short*)(smem + OFF_WBUF);  // slots: 0 Wr, 1 Wi, 2 -Wi
  unsigned short* act  = (unsigned short*)(smem + OFF_ACT);   // slots: 0 zr/ar, 1 zi/ai, 2 sr, 3 si
  float* fst  = (float*)(smem + OFF_FST);                     // slots: 0,1 gemm out; 2 gate
  float* pcur = (float*)(smem + OFF_PCUR);
  float* wmld = (float*)(smem + OFF_WM);
  float* cum  = (float*)(smem + OFF_CUM);
  float* wrow = (float*)(smem + OFF_WROW);
  float* scv  = (float*)(smem + OFF_SC);
  float* shv  = (float*)(smem + OFF_SH);
  float* mbv  = (float*)(smem + OFF_MB);
  float* gbv  = (float*)(smem + OFF_GB);
  float* hwv  = (float*)(smem + OFF_HW);
  float* swv  = (float*)(smem + OFF_SW);

  const int tid  = threadIdx.x;
  const int wave = tid >> 5;
  const int lane = tid & 31;
  const int m0   = blockIdx.x * TB;
  const float hbv = hb[0];
  const float sb0 = sb[0], sb1 = sb[1], sb2 = sb[2];

  // ---------------- per-launch init (deterministic) ----------------
  // bf16 weight tiles in workspace (all blocks write identical bytes).
  conv_ws   (wbf + 0*WTILE,               lwr, Dd,   tid);
  conv_ws_pm(wbf + 1*WTILE, wbf+2*WTILE,  lwi, Dd,   tid);
  conv_ws   (wbf + 3*WTILE,               vwr, Dd,   tid);
  conv_ws_pm(wbf + 4*WTILE, wbf+5*WTILE,  vwi, Dd,   tid);
  conv_ws   (wbf + 6*WTILE,               owr, Dd,   tid);
  conv_ws_pm(wbf + 7*WTILE, wbf+8*WTILE,  owi, Dd,   tid);
  for (int j = 0; j < 4; ++j) conv_ws(wbf + (9+j)*WTILE, gw + j*Dd, 4*Dd, tid);

  for (int i = tid; i < Dd; i += NTH) { scv[i]=nsc[i]; shv[i]=nsh[i]; mbv[i]=mbi[i]; gbv[i]=gb[i]; }
  for (int i = tid; i < 2*Dd; i += NTH)   hwv[i] = hw[i];
  for (int i = tid; i < 3*2*Dd; i += NTH) swv[i] = sw[i];
  if (tid < TB) cum[tid] = 0.0f;
  for (int e = tid; e < TB*Dd; e += NTH) {
    act[0*TB*Dd + e] = f2bf(zr_in[m0*Dd + e]);
    act[1*TB*Dd + e] = f2bf(zi_in[m0*Dd + e]);
    waccr[m0*Dd + e] = 0.0f;
    wacci[m0*Dd + e] = 0.0f;
  }
  for (int e = tid; e < TB*Ss; e += NTH) pcur[e] = ptr_in[m0*Ss + e];
  {
    // copy this block's stack-memory tile into workspace (L2-resident, 1.5 MB)
    const float4* src = (const float4*)(mem_in + (size_t)m0 * Ss * 2 * Dd);
    float4*       dst = (float4*)(wmem + (size_t)m0 * Ss * 2 * Dd);
    for (int i = tid; i < (TB*Ss*2*Dd)/4; i += NTH) dst[i] = src[i];
  }
  __threadfence();     // make bf16 weight tiles visible before async reads
  __syncthreads();

  for (int it = 0; it < DEPTH; ++it) {
    // ---- stage 1: complex linear (pr, pi) ----
    async_copy(wbuf, wbf + 0*WTILE, 3*WTILE*2, tid);   // Wr,Wi,-Wi in one DMA burst
    async_fence();
    __syncthreads();
    prefetch_bf(wbf + 3*WTILE, 3*WTILE, tid);
    cgemm(act, act + TB*Dd, wbuf, wbuf + WTILE, wbuf + 2*WTILE,
          fst, fst + TB*Dd, wave, lane);
    __syncthreads();

    // ---- stage 2: magnitude layernorm + phase + modReLU -> act0/1 ----
    for (int rr = 0; rr < 8; ++rr) {
      const int r = wave * 8 + rr;
      float prv[4], piv[4], mg[4];
      float s = 0.0f;
#pragma unroll
      for (int q = 0; q < 4; ++q) {
        const int c = lane + q * 32;
        prv[q] = fst[r*Dd + c];
        piv[q] = fst[TB*Dd + r*Dd + c];
        mg[q]  = sqrtf(prv[q]*prv[q] + piv[q]*piv[q]) + EPSV;
        s += mg[q];
      }
      for (int mk = 16; mk >= 1; mk >>= 1) s += __shfl_xor(s, mk, 32);
      const float mean = s * (1.0f / 128.0f);
      float vv = 0.0f;
#pragma unroll
      for (int q = 0; q < 4; ++q) { const float d = mg[q] - mean; vv += d*d; }
      for (int mk = 16; mk >= 1; mk >>= 1) vv += __shfl_xor(vv, mk, 32);
      const float rstd = rsqrtf(vv * (1.0f / 127.0f) + EPSV);
#pragma unroll
      for (int q = 0; q < 4; ++q) {
        const int c = lane + q * 32;
        const float nm  = (mg[q] - mean) * rstd * scv[c] + shv[c];
        const float inv = 1.0f / (mg[q] - EPSV + 1e-30f);   // cos=pr/|p|, sin=pi/|p|
        const float zr2 = nm * prv[q] * inv;
        const float zi2 = nm * piv[q] * inv;
        const float n2  = fabsf(nm) + EPSV;
        float t = n2 + mbv[c]; t = t > 0.0f ? t : 0.0f;
        const float sf = t / n2;
        act[r*Dd + c]           = f2bf(zr2 * sf);
        act[TB*Dd + r*Dd + c]   = f2bf(zi2 * sf);
      }
    }
    __syncthreads();

    // ---- stage 3: complex v-projection ----
    async_copy(wbuf, wbf + 3*WTILE, 3*WTILE*2, tid);
    async_fence();
    __syncthreads();
    prefetch_bf(wbf + 6*WTILE, 3*WTILE, tid);
    cgemm(act, act + TB*Dd, wbuf, wbuf + WTILE, wbuf + 2*WTILE,
          fst, fst + TB*Dd, wave, lane);
    __syncthreads();
    for (int e = tid; e < TB*Dd; e += NTH) {
      act[e]          = f2bf(fst[e]);
      act[TB*Dd + e]  = f2bf(fst[TB*Dd + e]);
    }
    __syncthreads();

    // ---- stage 4: complex o-projection -> ar, ai in fst0/1 ----
    async_copy(wbuf, wbf + 6*WTILE, 3*WTILE*2, tid);
    async_fence();
    __syncthreads();
    prefetch_bf(wbf + 9*WTILE, 4*WTILE, tid);
    cgemm(act, act + TB*Dd, wbuf, wbuf + WTILE, wbuf + 2*WTILE,
          fst, fst + TB*Dd, wave, lane);
    __syncthreads();

    // ---- stage 5: ar/ai -> bf16; per-row halt, ctrl softmax, stack pointer ----
    for (int e = tid; e < TB*Dd; e += NTH) {
      act[e]          = f2bf(fst[e]);
      act[TB*Dd + e]  = f2bf(fst[TB*Dd + e]);
    }
    if (tid < TB) {
      const int r = tid;
      const float* arp = fst + r*Dd;
      const float* aip = fst + TB*Dd + r*Dd;
      float lh = hbv, l0 = sb0, l1 = sb1, l2 = sb2;
      for (int k = 0; k < Dd; ++k) {
        const float a = arp[k], b = aip[k];
        lh += a * hwv[k]        + b * hwv[Dd + k];
        l0 += a * swv[k]        + b * swv[Dd + k];
        l1 += a * swv[2*Dd + k] + b * swv[3*Dd + k];
        l2 += a * swv[4*Dd + k] + b * swv[5*Dd + k];
      }
      const float halt = sigm(lh);
      const float cu = cum[r];
      const float running = (cu < THRv) ? 1.0f : 0.0f;
      const float wv = (((cu + halt) > THRv) ? (1.0f - cu) : halt) * running;
      cum[r]  = cu + wv;
      wrow[r] = wv;
      const float mx = fmaxf(l0, fmaxf(l1, l2));
      const float e0 = __expf(l0 - mx), e1 = __expf(l1 - mx), e2 = __expf(l2 - mx);
      const float isum = 1.0f / (e0 + e1 + e2);
      const float push = e0 * isum, pop = e1 * isum, noop = e2 * isum;
      float pold[Ss];
#pragma unroll
      for (int i = 0; i < Ss; ++i) pold[i] = pcur[r*Ss + i];
      float pn[Ss]; float psum = 0.0f;
#pragma unroll
      for (int i = 0; i < Ss; ++i) {
        const float pu = pold[(i + Ss - 1) % Ss];
        const float pd = pold[(i + 1) % Ss];
        const float np = push * pu + pop * pd + noop * pold[i];
        pn[i] = np; psum += np;
        wmld[r*Ss + i] = push * pu;
      }
      const float inv = 1.0f / (psum + EPSV);
#pragma unroll
      for (int i = 0; i < Ss; ++i) pcur[r*Ss + i] = pn[i] * inv;
    }
    __syncthreads();

    // ---- stage 6: halting accumulation + soft-stack memory update/read ----
    for (int e = tid; e < TB*Dd; e += NTH) {
      const int r = e >> 7;
      const float wv = wrow[r];
      waccr[m0*Dd + e] += wv * fst[e];
      wacci[m0*Dd + e] += wv * fst[TB*Dd + e];
    }
    {
      const int r  = tid >> 2;      // 4 threads per row
      const int q  = tid & 3;
      const int cb0 = q * 64;       // 64 of 256 columns
      const size_t gbase = (size_t)(m0 + r) * Ss * 2 * Dd;
      for (int ch = 0; ch < 4; ++ch) {
        const int cb = cb0 + ch * 16;
        float zf[16];
#pragma unroll
        for (int t = 0; t < 16; ++t) {
          const int c = cb + t;
          zf[t] = (c < Dd) ? fst[r*Dd + c] : fst[TB*Dd + r*Dd + (c - Dd)];
        }
        float racc[16];
#pragma unroll
        for (int t = 0; t < 16; ++t) racc[t] = 0.0f;
        for (int i = 0; i < Ss; ++i) {
          const float wmv = wmld[r*Ss + i];
          const float np  = pcur[r*Ss + i];
          float4* mp = (float4*)(wmem + gbase + (size_t)i * 2 * Dd + cb);
#pragma unroll
          for (int t4 = 0; t4 < 4; ++t4) {
            float4 v = mp[t4];
            v.x += wmv * (zf[t4*4+0] - v.x);
            v.y += wmv * (zf[t4*4+1] - v.y);
            v.z += wmv * (zf[t4*4+2] - v.z);
            v.w += wmv * (zf[t4*4+3] - v.w);
            mp[t4] = v;
            racc[t4*4+0] += v.x * np; racc[t4*4+1] += v.y * np;
            racc[t4*4+2] += v.z * np; racc[t4*4+3] += v.w * np;
          }
        }
#pragma unroll
        for (int t = 0; t < 16; ++t) {
          const int c = cb + t;
          if (c < Dd) act[2*TB*Dd + r*Dd + c]        = f2bf(racc[t]);
          else        act[3*TB*Dd + r*Dd + (c - Dd)] = f2bf(racc[t]);
        }
      }
    }
    __syncthreads();

    // ---- stage 7: gate GEMM: [ar|ai|sr|si](64x512) @ gate_w.T -> g in fst2 ----
    {
      const v8f zero8 = {0,0,0,0,0,0,0,0};
      v8f accg[4];
#pragma unroll
      for (int nt = 0; nt < 4; ++nt) accg[nt] = zero8;
      const int strip = wave >> 1, nbase = (wave & 1) * 4;
      for (int j = 0; j < 4; ++j) {
        unsigned short* wsl = wbuf + (j & 1) * WTILE;
        async_copy(wsl, wbf + (9 + j) * WTILE, WTILE * 2, tid);
        async_fence();
        __syncthreads();
        const unsigned short* ab = act + j * TB * Dd + strip * 16 * Dd;
        v16bf fa[4];
#pragma unroll
        for (int kk = 0; kk < 4; ++kk) fa[kk] = fragA(ab, lane, kk * 32);
        for (int nt = 0; nt < 4; ++nt) {
#pragma unroll
          for (int kk = 0; kk < 4; ++kk) {
            const v16bf fb = fragB(wsl, lane, nbase + nt, kk * 32);
            accg[nt] = wmma_bf(fa[kk], fb, accg[nt]);
          }
        }
        __syncthreads();
      }
      const int rbase = strip * 16 + 8 * (lane >> 4);
      for (int nt = 0; nt < 4; ++nt) {
        const int col = (nbase + nt) * 16 + (lane & 15);
        const float gbc = gbv[col];
#pragma unroll
        for (int v = 0; v < 8; ++v)
          fst[2*TB*Dd + (rbase + v)*Dd + col] = sigm(accg[nt][v] + gbc);
      }
    }
    __syncthreads();

    // ---- stage 8: gated recombination -> next zr, zi (bf16 in act0/1) ----
    for (int e = tid; e < TB*Dd; e += NTH) {
      const float g  = fst[2*TB*Dd + e];
      const float a  = fst[e];
      const float b  = fst[TB*Dd + e];
      const float sr = bf2f(act[2*TB*Dd + e]);
      const float si = bf2f(act[3*TB*Dd + e]);
      const float zr = (1.0f - g)*a + g*b + g*sr - g*si;
      const float zi = (1.0f - g)*b - g*a + g*si + g*sr;
      act[e]         = f2bf(zr);
      act[TB*Dd + e] = f2bf(zi);
    }
    __syncthreads();
  }

  // ---------------- emit output: concat[accr, acci] (B x 256) ----------------
  for (int e = tid; e < TB * 2 * Dd; e += NTH) {
    const int r = e >> 8, c = e & 255;
    const float v = (c < Dd) ? waccr[(m0 + r)*Dd + c]
                             : wacci[(m0 + r)*Dd + (c - Dd)];
    out[(size_t)(m0 + r) * 2 * Dd + c] = v;
  }
}

extern "C" void kernel_launch(void* const* d_in, const int* in_sizes, int n_in,
                              void* d_out, int out_size, void* d_ws, size_t ws_size,
                              hipStream_t stream) {
  const float* zr  = (const float*)d_in[0];
  const float* zi  = (const float*)d_in[1];
  const float* mem = (const float*)d_in[2];
  const float* ptr = (const float*)d_in[3];
  const float* lwr = (const float*)d_in[4];
  const float* lwi = (const float*)d_in[5];
  const float* nsc = (const float*)d_in[6];
  const float* nsh = (const float*)d_in[7];
  const float* mbi = (const float*)d_in[8];
  const float* vwr = (const float*)d_in[9];
  const float* vwi = (const float*)d_in[10];
  const float* owr = (const float*)d_in[11];
  const float* owi = (const float*)d_in[12];
  const float* hw  = (const float*)d_in[13];
  const float* hb  = (const float*)d_in[14];
  const float* sw  = (const float*)d_in[15];
  const float* sb  = (const float*)d_in[16];
  const float* gw  = (const float*)d_in[17];
  const float* gb  = (const float*)d_in[18];
  float* out = (float*)d_out;

  float* ws     = (float*)d_ws;
  float* wmem   = ws;                                       // B*S*2D f32 (100 MB, L2-resident)
  float* waccr  = ws + (size_t)Bsz * Ss * 2 * Dd;           // B*D
  float* wacci  = waccr + (size_t)Bsz * Dd;                 // B*D
  unsigned short* wbf = (unsigned short*)(wacci + (size_t)Bsz * Dd);  // 13 bf16 weight tiles

  (void)hipFuncSetAttribute((const void*)ubercrsn_kernel,
                            hipFuncAttributeMaxDynamicSharedMemorySize, SMEM_SZ);

  ubercrsn_kernel<<<NBLK, NTH, SMEM_SZ, stream>>>(
      zr, zi, mem, ptr, lwr, lwi, nsc, nsh, mbi, vwr, vwi, owr, owi,
      hw, hb, sw, sb, gw, gb, out, wmem, waccr, wacci, wbf);
}
